// VectorQuantizer_85452669321717
// MI455X (gfx1250) — matrix operand order repaired and checked by reading, not compiled, not run on hardware
//
#include <hip/hip_runtime.h>

// ---------------------------------------------------------------------------
// VQ-VAE vector quantizer for MI455X (gfx1250, wave32, WMMA).
// Core: fused bf16 WMMA GEMM (32768x8192x256) + register-resident argmax,
// never materializing the 1 GB distance matrix.  B tiles staged into
// double-buffered LDS with GLOBAL_LOAD_ASYNC_TO_LDS (ASYNCcnt) when the
// toolchain exposes the builtin.
// ---------------------------------------------------------------------------

#define K_DIM   256
#define N_CODES 8192
#define L_DIM   2048
#define B_DIM   16
#define C_DIM   256
#define BL_ROWS (B_DIM * L_DIM)   // 32768
#define BETA_F  0.1f
#define NTILE   32                // codes staged per iteration
#define NITER   (N_CODES / NTILE) // 256

typedef __attribute__((ext_vector_type(16))) __bf16      v16bf;
typedef __attribute__((ext_vector_type(8)))  float       v8f;
typedef __attribute__((ext_vector_type(4)))  unsigned int u32x4;
typedef __attribute__((ext_vector_type(4)))  int          i32x4;

union BfFrag { u32x4 q[2]; v16bf v; };

#if __has_builtin(__builtin_amdgcn_global_load_async_to_lds_b128)
#define HAVE_ASYNC_LDS 1
typedef __attribute__((address_space(1))) i32x4 gi32x4;   // global (AS1)
typedef __attribute__((address_space(3))) i32x4 li32x4;   // LDS (AS3)
__device__ __forceinline__ void async_copy16(const void* g, void* l) {
  __builtin_amdgcn_global_load_async_to_lds_b128(
      (gi32x4*)(unsigned long long)g,
      (li32x4*)(unsigned int)(unsigned long long)l, 0, 0);
}
#else
#define HAVE_ASYNC_LDS 0
#endif

__device__ __forceinline__ void wait_async_lds() {
#if HAVE_ASYNC_LDS
#if __has_builtin(__builtin_amdgcn_s_wait_asynccnt)
  __builtin_amdgcn_s_wait_asynccnt(0);
#else
  asm volatile("s_wait_asynccnt 0" ::: "memory");
#endif
#endif
}

__device__ __forceinline__ unsigned short f2bf_bits(float f) {
  union { float f; unsigned u; } v; v.f = f;
  unsigned u = v.u;
  unsigned r = u + 0x7FFFu + ((u >> 16) & 1u);   // round-to-nearest-even
  return (unsigned short)(r >> 16);
}

// ---------------------------------------------------------------------------
// 1) codebook -> bf16 copy + |e|^2 per code.  One wave32 per codebook row.
// ---------------------------------------------------------------------------
__global__ __launch_bounds__(256) void vq_prep_codebook(
    const float* __restrict__ cb, unsigned short* __restrict__ cb_bf,
    float* __restrict__ cnorm) {
  int tid = threadIdx.x;
  int wave = tid >> 5, lane = tid & 31;
  int n = blockIdx.x * 8 + wave;
  const float* rp = cb + (size_t)n * K_DIM + lane * 8;
  unsigned short* op = cb_bf + (size_t)n * K_DIM + lane * 8;
  float s = 0.f;
#pragma unroll
  for (int i = 0; i < 8; ++i) {
    float x = rp[i];
    s += x * x;
    op[i] = f2bf_bits(x);
  }
#pragma unroll
  for (int off = 16; off >= 1; off >>= 1) s += __shfl_xor(s, off, 32);
  if (lane == 0) cnorm[n] = s;
}

// ---------------------------------------------------------------------------
// 2) z (B,C,L) f32 -> zf (B*L, C) bf16, LDS-tiled transpose, coalesced.
// ---------------------------------------------------------------------------
__global__ __launch_bounds__(256) void vq_prep_z(
    const float* __restrict__ z, unsigned short* __restrict__ zf) {
  __shared__ float s[32][33];
  int b = blockIdx.z, c0 = blockIdx.y * 32, l0 = blockIdx.x * 32;
  int tx = threadIdx.x, ty = threadIdx.y;
#pragma unroll
  for (int j = 0; j < 4; ++j) {
    int c = c0 + ty + j * 8;                         // s[c_local][l_local]
    s[ty + j * 8][tx] = z[((size_t)b * C_DIM + c) * L_DIM + l0 + tx];
  }
  __syncthreads();
#pragma unroll
  for (int j = 0; j < 4; ++j) {
    int l = l0 + ty + j * 8;
    zf[((size_t)b * L_DIM + l) * K_DIM + c0 + tx] = f2bf_bits(s[tx][ty + j * 8]);
  }
}

// ---------------------------------------------------------------------------
// 3) Fused GEMM + argmax.  argmin(|z|^2+|e|^2-2ze) == argmax(2ze-|e|^2).
//    256 blocks x 8 waves; wave owns a 16-row A tile in registers (all K).
//    32-code B tiles double-buffered in LDS (async copy overlaps the
//    16-WMMA compute); 1 workgroup barrier per 32 codes.
// ---------------------------------------------------------------------------
__global__ __launch_bounds__(256) void vq_gemm_argmin(
    const unsigned short* __restrict__ zf,
    const unsigned short* __restrict__ cb_bf,
    const float* __restrict__ cnorm,
    int* __restrict__ inds_i, float* __restrict__ inds_f) {
  __shared__ __align__(16) unsigned short Btile[2][NTILE * K_DIM];  // 2 x 16 KB
  __shared__ float Cn[2][NTILE];

  int tid = threadIdx.x;
  int wave = tid >> 5, lane = tid & 31;
  int row_base = blockIdx.x * 128 + wave * 16;
  int mh = lane >> 4;          // lane-half: selects K sub-range per ISA layout
  int ml = lane & 15;          // row (A) / column (B,C) index

  // A fragments: 16-bit A 16x32 layout -> lane half h holds K[h*8 .. h*8+7]
  // and K[h*8+16 .. h*8+23] of each 32-wide K step (two 16B loads).
  BfFrag a[8];
#pragma unroll
  for (int ks = 0; ks < 8; ++ks) {
    const unsigned short* ap =
        zf + (size_t)(row_base + ml) * K_DIM + ks * 32 + mh * 8;
    a[ks].q[0] = *(const u32x4*)(ap);
    a[ks].q[1] = *(const u32x4*)(ap + 16);
  }

  float best[8];
  int   bidx[8];
#pragma unroll
  for (int r = 0; r < 8; ++r) { best[r] = -3.4e38f; bidx[r] = 0; }

  // copy share: thread covers 64 B of one codebook row (4 x b128)
  int crow = tid >> 3;         // 0..31 code row within N-tile
  int cseg = tid & 7;          // 32-element (64 B) segment of that row

  // stage tile nt into buffer buf (async when available, else sync copy;
  // either way made visible by the next workgroup barrier)
  auto stage = [&](int nt, int buf) {
    const unsigned short* src =
        cb_bf + ((size_t)nt * NTILE + crow) * K_DIM + cseg * 32;
    unsigned short* dst = &Btile[buf][crow * K_DIM + cseg * 32];
#if HAVE_ASYNC_LDS
#pragma unroll
    for (int i = 0; i < 4; ++i) async_copy16(src + i * 8, dst + i * 8);
#else
    const u32x4* s4 = (const u32x4*)src;
    u32x4* d4 = (u32x4*)dst;
#pragma unroll
    for (int i = 0; i < 4; ++i) d4[i] = s4[i];
#endif
    if (tid < NTILE) Cn[buf][tid] = cnorm[nt * NTILE + tid];
    if (nt + 1 < NITER)   // global_prefetch_b8 for the tile after next
      __builtin_prefetch(cb_bf + ((size_t)(nt + 1) * NTILE + crow) * K_DIM + cseg * 32, 0, 3);
  };

  stage(0, 0);

  for (int nt = 0; nt < NITER; ++nt) {
    int buf = nt & 1;
    wait_async_lds();    // this wave's copies for tile nt are done
    __syncthreads();     // tile nt visible from all waves; prev reads retired
    if (nt + 1 < NITER) stage(nt + 1, buf ^ 1);

    v8f c0 = {}, c1 = {};
#pragma unroll
    for (int ks = 0; ks < 8; ++ks) {
      // B 32x16 layout (matches A): lane half h holds K[h*8 .. h*8+7] and
      // K[h*8+16 .. h*8+23] of this K-32 block, column = ml (two b128 ds loads).
      BfFrag b0, b1;
      const u32x4* bp0 =
          (const u32x4*)(&Btile[buf][ml * K_DIM + ks * 32 + mh * 8]);
      const u32x4* bp1 =
          (const u32x4*)(&Btile[buf][(16 + ml) * K_DIM + ks * 32 + mh * 8]);
      b0.q[0] = bp0[0]; b0.q[1] = bp0[2];
      b1.q[0] = bp1[0]; b1.q[1] = bp1[2];
      c0 = __builtin_amdgcn_wmma_f32_16x16x32_bf16(false, a[ks].v, false, b0.v,
                                                   (short)0, c0, false, false);
      c1 = __builtin_amdgcn_wmma_f32_16x16x32_bf16(false, a[ks].v, false, b1.v,
                                                   (short)0, c1, false, false);
    }

    float cn0 = Cn[buf][ml], cn1 = Cn[buf][16 + ml];
    int n0 = nt * NTILE + ml, n1 = n0 + 16;
    // C layout: VGPR r, lanes 0-15 -> row r; lanes 16-31 -> row r+8; col = ml
#pragma unroll
    for (int r = 0; r < 8; ++r) {
      float s0 = 2.0f * c0[r] - cn0;
      if (s0 > best[r]) { best[r] = s0; bidx[r] = n0; }
      float s1 = 2.0f * c1[r] - cn1;
      if (s1 > best[r]) { best[r] = s1; bidx[r] = n1; }
    }
  }

  // argmax across the 16 columns: reduce within each 16-lane half (same row)
#pragma unroll
  for (int r = 0; r < 8; ++r) {
    float bs = best[r]; int bi = bidx[r];
#pragma unroll
    for (int off = 8; off >= 1; off >>= 1) {
      float s2 = __shfl_xor(bs, off, 16);
      int   i2 = __shfl_xor(bi, off, 16);
      if (s2 > bs || (s2 == bs && i2 < bi)) { bs = s2; bi = i2; }  // first-match tie
    }
    if (ml == 0) {
      int row = row_base + r + mh * 8;
      inds_i[row] = bi;
      inds_f[row] = (float)bi;
    }
  }
}

// ---------------------------------------------------------------------------
// 4) z_q = codebook[inds] written back transposed to (B,C,L), fused loss acc.
// ---------------------------------------------------------------------------
__global__ __launch_bounds__(256) void vq_gather_loss(
    const float* __restrict__ cb, const int* __restrict__ inds_i,
    const float* __restrict__ z, float* __restrict__ zq,
    float* __restrict__ acc) {
  __shared__ float s[32][33];
  __shared__ int ilds[32];
  __shared__ float partial[8];
  int b = blockIdx.z, c0 = blockIdx.y * 32, l0 = blockIdx.x * 32;
  int tx = threadIdx.x, ty = threadIdx.y;
  int tid = ty * 32 + tx;
  if (ty == 0) ilds[tx] = inds_i[b * L_DIM + l0 + tx];
  __syncthreads();
#pragma unroll
  for (int j = 0; j < 4; ++j) {
    int ll = ty + j * 8;                             // s[l_local][c_local]
    s[ll][tx] = cb[(size_t)ilds[ll] * K_DIM + c0 + tx];
  }
  __syncthreads();
  float a = 0.f;
#pragma unroll
  for (int j = 0; j < 4; ++j) {
    int c = c0 + ty + j * 8;
    size_t o = ((size_t)b * C_DIM + c) * L_DIM + l0 + tx;
    float q = s[tx][ty + j * 8];
    float d = q - z[o];
    zq[o] = q;
    a += d * d;
  }
#pragma unroll
  for (int off = 16; off >= 1; off >>= 1) a += __shfl_xor(a, off, 32);
  int wave = tid >> 5, lane = tid & 31;
  if (lane == 0) partial[wave] = a;
  __syncthreads();
  if (tid == 0) {
    float t = 0.f;
#pragma unroll
    for (int i = 0; i < 8; ++i) t += partial[i];
    atomicAdd(acc, t);
  }
}

__global__ void vq_zero_acc(float* acc) { *acc = 0.f; }

__global__ void vq_finalize(const float* __restrict__ acc,
                            float* __restrict__ loss_out) {
  *loss_out = (1.0f + BETA_F) * (*acc) / (float)(B_DIM * C_DIM * L_DIM);
}

// ---------------------------------------------------------------------------
extern "C" void kernel_launch(void* const* d_in, const int* in_sizes, int n_in,
                              void* d_out, int out_size, void* d_ws, size_t ws_size,
                              hipStream_t stream) {
  const float* z  = (const float*)d_in[0];   // (16, 256, 2048) f32
  const float* cb = (const float*)d_in[1];   // (8192, 256) f32

  char* ws = (char*)d_ws;
  unsigned short* zf_bf = (unsigned short*)(ws);               // 16 MB
  unsigned short* cb_bf = (unsigned short*)(ws + 16777216);    // 4 MB
  float* cnorm  = (float*)(ws + 20971520);                     // 32 KB
  int*   inds_i = (int*)(ws + 21004288);                       // 128 KB
  float* acc    = (float*)(ws + 21135360);                     // 4 B

  float* zq     = (float*)d_out;                               // 8388608 f32
  float* loss   = (float*)d_out + 8388608;                     // 1 f32
  float* inds_f = (float*)d_out + 8388609;                     // 32768 f32

  vq_prep_codebook<<<N_CODES / 8, 256, 0, stream>>>(cb, cb_bf, cnorm);
  vq_prep_z<<<dim3(L_DIM / 32, C_DIM / 32, B_DIM), dim3(32, 8), 0, stream>>>(z, zf_bf);
  vq_gemm_argmin<<<BL_ROWS / 128, 256, 0, stream>>>(zf_bf, cb_bf, cnorm, inds_i, inds_f);
  vq_zero_acc<<<1, 1, 0, stream>>>(acc);
  vq_gather_loss<<<dim3(L_DIM / 32, C_DIM / 32, B_DIM), dim3(32, 8), 0, stream>>>(
      cb, inds_i, z, zq, acc);
  vq_finalize<<<1, 1, 0, stream>>>(acc, loss);
}
